// UResNet_60095182406077
// MI455X (gfx1250) — matrix operand order repaired and checked by reading, not compiled, not run on hardware
//
#include <hip/hip_runtime.h>
#include <hip/hip_bf16.h>

typedef _Float16 f16;
typedef __attribute__((ext_vector_type(16))) _Float16 v16h;
typedef __attribute__((ext_vector_type(8)))  _Float16 v8h;
typedef __attribute__((ext_vector_type(8)))  float    v8f;

#define TPB 256
#define SRED 512

__device__ __forceinline__ v8h zero8() {
  v8h z;
#pragma unroll
  for (int s = 0; s < 8; ++s) z[s] = (f16)0.f;
  return z;
}

// ---------------- elementwise / setup kernels ----------------

__global__ void k_zero(float* p, int n) {
  int i = blockIdx.x * blockDim.x + threadIdx.x;
  if (i < n) p[i] = 0.f;
}

__global__ void k_scatter(const int* __restrict__ coords, const float* __restrict__ feat,
                          float* __restrict__ dense, float* __restrict__ mask, int N, int G) {
  int n = blockIdx.x * blockDim.x + threadIdx.x;
  if (n >= N) return;
  int i0 = coords[3 * n], i1 = coords[3 * n + 1], i2 = coords[3 * n + 2];
  int v = (i0 * G + i1) * G + i2;
  atomicAdd(&dense[v], feat[n]);
  mask[v] = 1.f;
}

// 3x3x3 SAME conv, 1 -> 16 channels, masked output (stem)
__global__ void k_conv_in(const float* __restrict__ dense, const float* __restrict__ W,
                          const float* __restrict__ mask, float* __restrict__ out, int G) {
  int v = blockIdx.x * blockDim.x + threadIdx.x;
  int V = G * G * G;
  if (v >= V) return;
  int d = v / (G * G), h = (v / G) % G, w = v % G;
  float acc[16];
#pragma unroll
  for (int c = 0; c < 16; ++c) acc[c] = 0.f;
  for (int kd = 0; kd < 3; ++kd)
    for (int kh = 0; kh < 3; ++kh)
      for (int kw = 0; kw < 3; ++kw) {
        int z = d + kd - 1, y = h + kh - 1, x = w + kw - 1;
        if ((unsigned)z < (unsigned)G && (unsigned)y < (unsigned)G && (unsigned)x < (unsigned)G) {
          float val = dense[(z * G + y) * G + x];
          const float* wp = W + ((kd * 3 + kh) * 3 + kw) * 16;
#pragma unroll
          for (int c = 0; c < 16; ++c) acc[c] += val * wp[c];
        }
      }
  float mk = mask[v];
#pragma unroll
  for (int c = 0; c < 16; ++c) out[(size_t)v * 16 + c] = acc[c] * mk;
}

// ---------------- masked batchnorm ----------------

__global__ void k_bn_stats(const float* __restrict__ x, const float* __restrict__ mask,
                           float* __restrict__ stats, int V, int C) {
  int c = blockIdx.x;
  float s = 0.f, s2 = 0.f, cm = 0.f;
  for (int v = threadIdx.x; v < V; v += blockDim.x) {
    float mk = mask[v];
    float val = x[(size_t)v * C + c] * mk;
    s += val; s2 += val * val; cm += mk;
  }
  __shared__ float r0[SRED], r1[SRED], r2[SRED];
  int t = threadIdx.x;
  r0[t] = s; r1[t] = s2; r2[t] = cm;
  __syncthreads();
  for (int off = SRED / 2; off > 0; off >>= 1) {
    if (t < off) { r0[t] += r0[t + off]; r1[t] += r1[t + off]; r2[t] += r2[t + off]; }
    __syncthreads();
  }
  if (t == 0) {
    stats[c] = r0[0];
    stats[C + c] = r1[0];
    if (c == 0) stats[2 * C] = r2[0];
  }
}

__global__ void k_bn_apply(const float* __restrict__ x, const float* __restrict__ mask,
                           const float* __restrict__ stats, const float* __restrict__ gg,
                           const float* __restrict__ bb, f16* __restrict__ out, int V, int C) {
  long long i = (long long)blockIdx.x * blockDim.x + threadIdx.x;
  long long n = (long long)V * C;
  if (i >= n) return;
  int c = (int)(i % C);
  int v = (int)(i / C);
  float cnt = fmaxf(stats[2 * C], 1.f);
  float mean = stats[c] / cnt;
  float var = stats[C + c] / cnt - mean * mean;
  float y = (x[i] - mean) * rsqrtf(var + 1e-4f) * gg[c] + bb[c];
  y = fmaxf(y, 0.f) * mask[v];
  out[i] = (f16)y;
}

// f32 -> f16, plain layout (down/up conv weights)
__global__ void k_cvt(const float* __restrict__ in, f16* __restrict__ out, int n) {
  int i = blockIdx.x * blockDim.x + threadIdx.x;
  if (i < n) out[i] = (f16)in[i];
}

// f32 [k, co] -> f16 transposed [co, kpad] with zero-padded K tail
__global__ void k_cvt_tr(const float* __restrict__ in, f16* __restrict__ out,
                         int Ktot, int Kpad, int Cout) {
  long long i = (long long)blockIdx.x * blockDim.x + threadIdx.x;
  if (i >= (long long)Cout * Kpad) return;
  int co = (int)(i / Kpad);
  int k = (int)(i - (long long)co * Kpad);
  out[i] = (k < Ktot) ? (f16)in[(size_t)k * Cout + co] : (f16)0.f;
}

// ---------------- WMMA implicit-GEMM 3x3x3 SAME conv ----------------
// Shared prologue/epilogue logic for all variants:
//   one wave -> 16-voxel x 16-outC tile on v_wmma_f32_16x16x32_f16;
//   A run0 -> slots 0..7 (K=k0+8g+s), run1 -> slots 8..15 (K=k0+16+8g+s);
//   B lane n holds K=k0+8g+s / k0+16+8g+s, same as A, in the [co, Kpad] transposed weights.

#define CONV3_PROLOGUE                                                         \
  int wave = blockIdx.x * (blockDim.x >> 5) + (threadIdx.x >> 5);              \
  int tilesV = (V + 15) >> 4;                                                  \
  int total = tilesV * (Cout >> 4);                                            \
  if (wave >= total) return;                                                   \
  int cot = wave / tilesV;                                                     \
  int tv = wave - cot * tilesV;                                                \
  int lane = threadIdx.x & 31;                                                 \
  int m = lane & 15, g = lane >> 4;                                            \
  int vA = tv * 16 + m;                                                        \
  bool vok = vA < V;                                                           \
  int dA = 0, hA = 0, wA = 0;                                                  \
  if (vok) { dA = vA / (G * G); hA = (vA / G) % G; wA = vA % G; }              \
  int n = cot * 16 + m;                                                        \
  v8f acc = {0.f, 0.f, 0.f, 0.f, 0.f, 0.f, 0.f, 0.f};

#define CONV3_EPILOGUE                                                         \
  _Pragma("unroll")                                                            \
  for (int r = 0; r < 8; ++r) {                                                \
    int vm = tv * 16 + r + 8 * g;                                              \
    if (vm < V) {                                                              \
      float val = acc[r] * mask[vm];                                           \
      size_t o = (size_t)vm * Cout + n;                                        \
      if (accum) out[o] += val; else out[o] = val;                             \
    }                                                                          \
  }

__device__ __forceinline__ v8f wmma_step(v8h a0, v8h a1, v8h b0, v8h b1, v8f acc) {
  v16h A, B;
#pragma unroll
  for (int s = 0; s < 8; ++s) {
    A[s] = a0[s]; A[8 + s] = a1[s];
    B[s] = b0[s]; B[8 + s] = b1[s];
  }
  return __builtin_amdgcn_wmma_f32_16x16x32_f16(false, A, false, B, (short)0, acc, false, false);
}

// CIN == 16: chunk j covers taps 2j (run0) and 2j+1 (run1), channel offset 8g.
// Fully unrolled: tap -> (kd,kh,kw) decode is compile-time.
__global__ void k_conv3_c16(const f16* __restrict__ x, const f16* __restrict__ w,
                            const float* __restrict__ mask, float* __restrict__ out,
                            int G, int V, int Cout, int Kpad, int accum) {
  CONV3_PROLOGUE
  const f16* wn = w + (size_t)n * Kpad + 8 * g;
#pragma unroll
  for (int j = 0; j < 14; ++j) {
    int tA = 2 * j, tB = 2 * j + 1;
    v8h a0 = zero8(), a1 = zero8();
    {
      int kd = tA / 9, kh = (tA / 3) % 3, kw = tA % 3;
      int zi = dA + kd - 1, yi = hA + kh - 1, xi = wA + kw - 1;
      if (vok && (unsigned)zi < (unsigned)G && (unsigned)yi < (unsigned)G && (unsigned)xi < (unsigned)G)
        a0 = *(const v8h*)(x + ((size_t)((long long)(zi * G + yi) * G + xi)) * 16 + 8 * g);
    }
    if (tB < 27) {
      int kd = tB / 9, kh = (tB / 3) % 3, kw = tB % 3;
      int zi = dA + kd - 1, yi = hA + kh - 1, xi = wA + kw - 1;
      if (vok && (unsigned)zi < (unsigned)G && (unsigned)yi < (unsigned)G && (unsigned)xi < (unsigned)G)
        a1 = *(const v8h*)(x + ((size_t)((long long)(zi * G + yi) * G + xi)) * 16 + 8 * g);
    }
    v8h b0 = *(const v8h*)(wn + 32 * j);
    v8h b1 = *(const v8h*)(wn + 32 * j + 16);
    acc = wmma_step(a0, a1, b0, b1, acc);
  }
  CONV3_EPILOGUE
}

// CIN % 32 == 0 (32/64/96/128): chunks never straddle taps. Nested kd/kh/kw
// loops: bounds + voxel base once per tap, unrolled channel loop, linear B walk.
template <int CIN>
__global__ void k_conv3_m32(const f16* __restrict__ x, const f16* __restrict__ w,
                            const float* __restrict__ mask, float* __restrict__ out,
                            int G, int V, int Cout, int Kpad, int accum) {
  CONV3_PROLOGUE
  const f16* wp = w + (size_t)n * Kpad + 8 * g;
  for (int kd = 0; kd < 3; ++kd) {
    int zi = dA + kd - 1;
    for (int kh = 0; kh < 3; ++kh) {
      int yi = hA + kh - 1;
#pragma unroll
      for (int kw = 0; kw < 3; ++kw) {
        int xi = wA + kw - 1;
        bool inb = vok && (unsigned)zi < (unsigned)G && (unsigned)yi < (unsigned)G &&
                   (unsigned)xi < (unsigned)G;
        const f16* xb = x + (long long)((long long)(zi * G + yi) * G + xi) * CIN + 8 * g;
#pragma unroll
        for (int c0 = 0; c0 < CIN; c0 += 32) {
          v8h a0 = zero8(), a1 = zero8();
          if (inb) {
            a0 = *(const v8h*)(xb + c0);
            a1 = *(const v8h*)(xb + c0 + 16);
          }
          v8h b0 = *(const v8h*)(wp);
          v8h b1 = *(const v8h*)(wp + 16);
          wp += 32;
          acc = wmma_step(a0, a1, b0, b1, acc);
        }
      }
    }
  }
  CONV3_EPILOGUE
}

// Generic fallback (CIN = 48 / 80, only used at tiny scales 2 and 4)
template <int CIN>
__global__ void k_conv3_gen(const f16* __restrict__ x, const f16* __restrict__ w,
                            const float* __restrict__ mask, float* __restrict__ out,
                            int G, int V, int Cout, int Kpad, int accum) {
  constexpr int Ktot = 27 * CIN;
  CONV3_PROLOGUE
  const f16* wn = w + (size_t)n * Kpad + 8 * g;
  for (int k0 = 0; k0 < Ktot; k0 += 32) {
    v8h a0 = zero8(), a1 = zero8();
    {
      int kstart = k0 + 8 * g;
      if (vok && kstart < Ktot) {
        int tap = kstart / CIN, c = kstart - tap * CIN;
        int kd = tap / 9, kh = (tap / 3) % 3, kw = tap % 3;
        int zi = dA + kd - 1, yi = hA + kh - 1, xi = wA + kw - 1;
        if ((unsigned)zi < (unsigned)G && (unsigned)yi < (unsigned)G && (unsigned)xi < (unsigned)G)
          a0 = *(const v8h*)(x + ((size_t)(zi * G + yi) * G + xi) * CIN + c);
      }
    }
    {
      int kstart = k0 + 16 + 8 * g;
      if (vok && kstart < Ktot) {
        int tap = kstart / CIN, c = kstart - tap * CIN;
        int kd = tap / 9, kh = (tap / 3) % 3, kw = tap % 3;
        int zi = dA + kd - 1, yi = hA + kh - 1, xi = wA + kw - 1;
        if ((unsigned)zi < (unsigned)G && (unsigned)yi < (unsigned)G && (unsigned)xi < (unsigned)G)
          a1 = *(const v8h*)(x + ((size_t)(zi * G + yi) * G + xi) * CIN + c);
      }
    }
    v8h b0 = *(const v8h*)(wn + k0);
    v8h b1 = *(const v8h*)(wn + k0 + 16);
    acc = wmma_step(a0, a1, b0, b1, acc);
  }
  CONV3_EPILOGUE
}

// ---------------- small direct kernels ----------------

__global__ void k_nin(const float* __restrict__ x, const float* __restrict__ W,
                      float* __restrict__ out, int V, int Cin, int Cout) {
  long long i = (long long)blockIdx.x * blockDim.x + threadIdx.x;
  if (i >= (long long)V * Cout) return;
  int co = (int)(i % Cout);
  int v = (int)(i / Cout);
  const float* xr = x + (size_t)v * Cin;
  float acc = 0.f;
  for (int ci = 0; ci < Cin; ++ci) acc += xr[ci] * W[(size_t)ci * Cout + co];
  out[i] = acc;
}

__global__ void k_downconv(const f16* __restrict__ x, const f16* __restrict__ w,
                           float* __restrict__ out, int Gin, int Cin, int Cout) {
  int Go = Gin >> 1;
  long long Vo = (long long)Go * Go * Go;
  long long i = (long long)blockIdx.x * blockDim.x + threadIdx.x;
  if (i >= Vo * Cout) return;
  int co = (int)(i % Cout);
  int vo = (int)(i / Cout);
  int d = vo / (Go * Go), h = (vo / Go) % Go, ww = vo % Go;
  float acc = 0.f;
  for (int a = 0; a < 2; ++a)
    for (int b = 0; b < 2; ++b)
      for (int c2 = 0; c2 < 2; ++c2) {
        const f16* xr = x + ((size_t)((2 * d + a) * Gin + (2 * h + b)) * Gin + (2 * ww + c2)) * Cin;
        const f16* wr = w + (size_t)((a * 2 + b) * 2 + c2) * Cin * Cout + co;
        for (int ci = 0; ci < Cin; ++ci) acc += (float)xr[ci] * (float)wr[(size_t)ci * Cout];
      }
  out[(size_t)vo * Cout + co] = acc;
}

__global__ void k_downpool(const float* __restrict__ mi, float* __restrict__ mo, int Go) {
  int vo = blockIdx.x * blockDim.x + threadIdx.x;
  int Vo = Go * Go * Go;
  if (vo >= Vo) return;
  int Gin = Go * 2;
  int d = vo / (Go * Go), h = (vo / Go) % Go, w = vo % Go;
  float m = 0.f;
  for (int a = 0; a < 2; ++a)
    for (int b = 0; b < 2; ++b)
      for (int c = 0; c < 2; ++c)
        m = fmaxf(m, mi[((2 * d + a) * Gin + (2 * h + b)) * Gin + (2 * w + c)]);
  mo[vo] = m;
}

__global__ void k_upconv(const f16* __restrict__ x, const f16* __restrict__ w,
                         const float* __restrict__ maskO, float* __restrict__ dst,
                         int Gout, int Cin, int Cout, int dstStride, int dstOff) {
  int Gin = Gout >> 1;
  long long Vo = (long long)Gout * Gout * Gout;
  long long i = (long long)blockIdx.x * blockDim.x + threadIdx.x;
  if (i >= Vo * Cout) return;
  int co = (int)(i % Cout);
  int vo = (int)(i / Cout);
  int d = vo / (Gout * Gout), h = (vo / Gout) % Gout, ww = vo % Gout;
  int tap = ((d & 1) * 2 + (h & 1)) * 2 + (ww & 1);
  const f16* xr = x + ((size_t)((d >> 1) * Gin + (h >> 1)) * Gin + (ww >> 1)) * Cin;
  const f16* wr = w + (size_t)tap * Cin * Cout + co;
  float acc = 0.f;
  for (int ci = 0; ci < Cin; ++ci) acc += (float)xr[ci] * (float)wr[(size_t)ci * Cout];
  dst[(size_t)vo * dstStride + dstOff + co] = acc * maskO[vo];
}

__global__ void k_copy_ch(const float* __restrict__ src, float* __restrict__ dst,
                          int V, int Cs, int Cd, int off) {
  long long i = (long long)blockIdx.x * blockDim.x + threadIdx.x;
  if (i >= (long long)V * Cs) return;
  int c = (int)(i % Cs);
  int v = (int)(i / Cs);
  dst[(size_t)v * Cd + off + c] = src[i];
}

__global__ void k_gather_linear(const f16* __restrict__ x, const int* __restrict__ coords,
                                const float* __restrict__ Wl, const float* __restrict__ bl,
                                float* __restrict__ out, int N, int G) {
  int n = blockIdx.x * blockDim.x + threadIdx.x;
  if (n >= N) return;
  int i0 = coords[3 * n], i1 = coords[3 * n + 1], i2 = coords[3 * n + 2];
  const f16* fr = x + ((size_t)(i0 * G + i1) * G + i2) * 16;
#pragma unroll
  for (int k = 0; k < 5; ++k) {
    float s = bl[k];
    for (int c = 0; c < 16; ++c) s += (float)fr[c] * Wl[c * 5 + k];
    out[(size_t)n * 5 + k] = s;
  }
}

// ---------------- host side ----------------

static inline int nblk(long long n) { return (int)((n + TPB - 1) / TPB); }
static inline int kpad32(int k) { return (k + 31) & ~31; }

static void launch_conv3(const f16* x, const f16* w, const float* mask, float* out,
                         int G, int V, int Cin, int Cout, int Kpad, int accum, hipStream_t s) {
  int tiles = ((V + 15) / 16) * (Cout / 16);
  const int wpb = 8;                         // 8 waves (256 threads) per block
  int blocks = (tiles + wpb - 1) / wpb;
  dim3 b(32 * wpb);
  switch (Cin) {
    case 16:  k_conv3_c16<<<blocks, b, 0, s>>>(x, w, mask, out, G, V, Cout, Kpad, accum); break;
    case 32:  k_conv3_m32<32><<<blocks, b, 0, s>>>(x, w, mask, out, G, V, Cout, Kpad, accum); break;
    case 64:  k_conv3_m32<64><<<blocks, b, 0, s>>>(x, w, mask, out, G, V, Cout, Kpad, accum); break;
    case 96:  k_conv3_m32<96><<<blocks, b, 0, s>>>(x, w, mask, out, G, V, Cout, Kpad, accum); break;
    case 128: k_conv3_m32<128><<<blocks, b, 0, s>>>(x, w, mask, out, G, V, Cout, Kpad, accum); break;
    case 48:  k_conv3_gen<48><<<blocks, b, 0, s>>>(x, w, mask, out, G, V, Cout, Kpad, accum); break;
    case 80:  k_conv3_gen<80><<<blocks, b, 0, s>>>(x, w, mask, out, G, V, Cout, Kpad, accum); break;
    default: break;
  }
}

extern "C" void kernel_launch(void* const* d_in, const int* in_sizes, int n_in,
                              void* d_out, int out_size, void* d_ws, size_t ws_size,
                              hipStream_t stream) {
  (void)n_in; (void)out_size; (void)ws_size;
  const int S = 5;
  const int NP[5] = {16, 32, 48, 64, 80};
  const int G[5] = {96, 48, 24, 12, 6};
  int V[5];
  for (int i = 0; i < 5; ++i) V[i] = G[i] * G[i] * G[i];
  const int N = in_sizes[0] / 3;

  // -------- unpack inputs (JAX pytree sorted-key flatten order) --------
  int p = 0;
  const int* coords = (const int*)d_in[p++];
  const float* feat = (const float*)d_in[p++];

  struct Blk { const float *W1, *W2, *bn1b, *bn1g, *bn2b, *bn2g, *nin; };
  struct Dec { Blk blk[2]; const float *upW, *upbnb, *upbng; } dec[4];
  struct Enc { Blk blk[2]; const float *dnW, *dnbnb, *dnbng; } enc[5];

  for (int d = 0; d < 4; ++d) {
    for (int r = 0; r < 2; ++r) {
      Blk& b = dec[d].blk[r];
      b.W1 = (const float*)d_in[p++];
      b.W2 = (const float*)d_in[p++];
      b.bn1b = (const float*)d_in[p++];
      b.bn1g = (const float*)d_in[p++];
      b.bn2b = (const float*)d_in[p++];
      b.bn2g = (const float*)d_in[p++];
      b.nin = (r == 0) ? (const float*)d_in[p++] : nullptr;
    }
    dec[d].upW = (const float*)d_in[p++];
    dec[d].upbnb = (const float*)d_in[p++];
    dec[d].upbng = (const float*)d_in[p++];
  }
  for (int i = 0; i < 5; ++i) {
    for (int r = 0; r < 2; ++r) {
      Blk& b = enc[i].blk[r];
      b.W1 = (const float*)d_in[p++];
      b.W2 = (const float*)d_in[p++];
      b.bn1b = (const float*)d_in[p++];
      b.bn1g = (const float*)d_in[p++];
      b.bn2b = (const float*)d_in[p++];
      b.bn2g = (const float*)d_in[p++];
      b.nin = nullptr;
    }
    if (i < 4) {
      enc[i].dnW = (const float*)d_in[p++];
      enc[i].dnbnb = (const float*)d_in[p++];
      enc[i].dnbng = (const float*)d_in[p++];
    } else {
      enc[i].dnW = nullptr; enc[i].dnbnb = nullptr; enc[i].dnbng = nullptr;
    }
  }
  const float* in_W = (const float*)d_in[p++];
  const float* lin_W = (const float*)d_in[p++];
  const float* lin_b = (const float*)d_in[p++];
  const float* out_bnb = (const float*)d_in[p++];
  const float* out_bng = (const float*)d_in[p++];

  // -------- workspace carve --------
  char* base = (char*)d_ws;
  size_t off = 0;
  auto carve = [&](size_t bytes) -> void* {
    void* r = base + off;
    off += (bytes + 255) & ~(size_t)255;
    return r;
  };
  float* mask[5];
  for (int i = 0; i < 5; ++i) mask[i] = (float*)carve((size_t)V[i] * 4);
  float* save[4];
  for (int i = 0; i < 4; ++i) save[i] = (float*)carve((size_t)V[i] * NP[i] * 4);
  size_t bigN = (size_t)V[0] * 32;
  float* bA = (float*)carve(bigN * 4);
  float* bB = (float*)carve(bigN * 4);
  float* bC = (float*)carve(bigN * 4);
  f16* H = (f16*)carve(bigN * 2);
  f16* Wh = (f16*)carve((size_t)232000 * 2);   // max Cout*Kpad = 64*3456 = 221184
  float* stats = (float*)carve((size_t)(2 * 160 + 1) * 4);

  // -------- voxelize + stem --------
  k_zero<<<nblk(V[0]), TPB, 0, stream>>>(bA, V[0]);
  k_zero<<<nblk(V[0]), TPB, 0, stream>>>(mask[0], V[0]);
  k_scatter<<<nblk(N), TPB, 0, stream>>>(coords, feat, bA, mask[0], N, G[0]);
  k_conv_in<<<nblk(V[0]), TPB, 0, stream>>>(bA, in_W, mask[0], bB, G[0]);

  float* x = bB;
  float* f1 = bA;
  float* f2 = bC;

  auto bn = [&](const float* xin, const float* mk, const float* gg, const float* bb,
                int Vv, int C) {
    k_bn_stats<<<C, SRED, 0, stream>>>(xin, mk, stats, Vv, C);
    k_bn_apply<<<nblk((long long)Vv * C), TPB, 0, stream>>>(xin, mk, stats, gg, bb, H, Vv, C);
  };

  auto doBlock = [&](const Blk& bp, int a, int b, const float* mk, int Gg, int Vv) {
    if (bp.nin)
      k_nin<<<nblk((long long)Vv * b), TPB, 0, stream>>>(x, bp.nin, f1, Vv, a, b);
    else
      hipMemcpyAsync(f1, x, (size_t)Vv * a * 4, hipMemcpyDeviceToDevice, stream);
    int Kt1 = 27 * a, Kp1 = kpad32(Kt1);
    bn(x, mk, bp.bn1g, bp.bn1b, Vv, a);
    k_cvt_tr<<<nblk((long long)b * Kp1), TPB, 0, stream>>>(bp.W1, Wh, Kt1, Kp1, b);
    launch_conv3(H, Wh, mk, f2, Gg, Vv, a, b, Kp1, 0, stream);
    int Kt2 = 27 * b, Kp2 = kpad32(Kt2);
    bn(f2, mk, bp.bn2g, bp.bn2b, Vv, b);
    k_cvt_tr<<<nblk((long long)b * Kp2), TPB, 0, stream>>>(bp.W2, Wh, Kt2, Kp2, b);
    launch_conv3(H, Wh, mk, f1, Gg, Vv, b, b, Kp2, 1, stream);
    float* t = x; x = f1; f1 = t;
  };

  // -------- encoder --------
  for (int i = 0; i < S; ++i) {
    for (int r = 0; r < 2; ++r) doBlock(enc[i].blk[r], NP[i], NP[i], mask[i], G[i], V[i]);
    if (i < 4) {
      hipMemcpyAsync(save[i], x, (size_t)V[i] * NP[i] * 4, hipMemcpyDeviceToDevice, stream);
      bn(x, mask[i], enc[i].dnbng, enc[i].dnbnb, V[i], NP[i]);
      k_cvt<<<nblk(8 * NP[i] * NP[i + 1]), TPB, 0, stream>>>(enc[i].dnW, Wh, 8 * NP[i] * NP[i + 1]);
      k_downconv<<<nblk((long long)V[i + 1] * NP[i + 1]), TPB, 0, stream>>>(H, Wh, f1, G[i], NP[i], NP[i + 1]);
      k_downpool<<<nblk(V[i + 1]), TPB, 0, stream>>>(mask[i], mask[i + 1], G[i + 1]);
      float* t = x; x = f1; f1 = t;
    }
  }

  // -------- decoder --------
  for (int d = 0; d < 4; ++d) {
    int i = 3 - d;
    int C = NP[i], Cup = NP[i + 1];
    bn(x, mask[i + 1], dec[d].upbng, dec[d].upbnb, V[i + 1], Cup);
    k_cvt<<<nblk(8 * Cup * C), TPB, 0, stream>>>(dec[d].upW, Wh, 8 * Cup * C);
    k_upconv<<<nblk((long long)V[i] * C), TPB, 0, stream>>>(H, Wh, mask[i], f1, G[i], Cup, C, 2 * C, C);
    k_copy_ch<<<nblk((long long)V[i] * C), TPB, 0, stream>>>(save[i], f1, V[i], C, 2 * C, 0);
    float* t = x; x = f1; f1 = t;
    doBlock(dec[d].blk[0], 2 * C, C, mask[i], G[i], V[i]);
    doBlock(dec[d].blk[1], C, C, mask[i], G[i], V[i]);
  }

  // -------- head --------
  bn(x, mask[0], out_bng, out_bnb, V[0], 16);
  k_gather_linear<<<nblk(N), TPB, 0, stream>>>(H, coords, lin_W, lin_b, (float*)d_out, N, G[0]);
}
